// MyVMLMFCellg2_67027259621390
// MI455X (gfx1250) — matrix operand order repaired and checked by reading, not compiled, not run on hardware
//
#include <hip/hip_runtime.h>
#include <stdint.h>

// MI455X / gfx1250, wave32. bf16 WMMA (f32 accumulate) for all GEMMs,
// async global->LDS staging of phase-2 weight tiles.
// B=4096, I=H=1024, R=64, G=2, Hg=512.

#define B_SZ 4096
#define H_SZ 1024
#define HG   512

typedef __attribute__((ext_vector_type(16))) __bf16 v16bf;
typedef __attribute__((ext_vector_type(8)))  float  v8f;

union BfVec {
  uint4          u[2];
  unsigned short s[16];
  v16bf          v;
};

__device__ __forceinline__ unsigned short f2bf(float f) {
  unsigned u = __builtin_bit_cast(unsigned, f);
  u += 0x7FFFu + ((u >> 16) & 1u);           // round-to-nearest-even
  return (unsigned short)(u >> 16);
}

// pack two floats into two bf16 (round-half-up bias + v_perm_b32 byte select)
__device__ __forceinline__ unsigned pack_bf(float lo, float hi) {
  unsigned ul = __builtin_bit_cast(unsigned, lo) + 0x8000u;
  unsigned uh = __builtin_bit_cast(unsigned, hi) + 0x8000u;
  return __builtin_amdgcn_perm(uh, ul, 0x07060302u);  // {uh[31:16], ul[31:16]}
}

// A operand (16-bit, 16x32): lane half h reads row M=lane&15:
// elements 0..7  = K [kbase + h*8, +8) ; elements 8..15 = K [kbase + 16 + h*8, +8)
__device__ __forceinline__ v16bf load_a_bf(const unsigned short* rowp, int half, int kbase) {
  BfVec r;
  r.u[0] = *(const uint4*)(rowp + kbase + half * 8);
  r.u[1] = *(const uint4*)(rowp + kbase + 16 + half * 8);
  return r.v;
}

__device__ __forceinline__ v16bf load_a_f32(const float* rowp, int half, int kbase) {
  const float* p0 = rowp + kbase + half * 8;
  const float* p1 = rowp + kbase + 16 + half * 8;
  float4 a0 = *(const float4*)(p0);
  float4 a1 = *(const float4*)(p0 + 4);
  float4 b0 = *(const float4*)(p1);
  float4 b1 = *(const float4*)(p1 + 4);
  BfVec r;
  r.u[0].x = pack_bf(a0.x, a0.y);
  r.u[0].y = pack_bf(a0.z, a0.w);
  r.u[0].z = pack_bf(a1.x, a1.y);
  r.u[0].w = pack_bf(a1.z, a1.w);
  r.u[1].x = pack_bf(b0.x, b0.y);
  r.u[1].y = pack_bf(b0.z, b0.w);
  r.u[1].z = pack_bf(b1.x, b1.y);
  r.u[1].w = pack_bf(b1.z, b1.w);
  return r.v;
}

// B operand (16-bit, 32x16): lane holds column N=lane&15;
// elements 0..7 = K [kbase + half*8, +8) ; elements 8..15 = K [kbase + 16 + half*8, +8). Rows stored [N][K].
__device__ __forceinline__ v16bf load_b_op(const unsigned short* rowp, int half, int kbase) {
  const unsigned short* p = rowp + kbase + half * 8;
  BfVec r;
  r.u[0] = *(const uint4*)(p);
  r.u[1] = *(const uint4*)(p + 16);
  return r.v;
}

__device__ __forceinline__ v8f wmma_bf16(v16bf a, v16bf b, v8f c) {
  return __builtin_amdgcn_wmma_f32_16x16x32_bf16(false, a, false, b, (short)0, c, false, false);
}

__device__ __forceinline__ void wait_asynccnt0() {
#if __has_builtin(__builtin_amdgcn_s_wait_asynccnt)
  __builtin_amdgcn_s_wait_asynccnt(0);
#else
  asm volatile("s_wait_asynccnt 0x0" ::: "memory");
#endif
}

// ---------------- prep: weight casts / transposes into [N][K] bf16 ----------
__global__ void prep_cast(const float* __restrict__ u_x, const float* __restrict__ u_h_0,
                          const float* __restrict__ u_h_1, const float* __restrict__ v_x,
                          const float* __restrict__ v_h_0, const float* __restrict__ v_h_1,
                          unsigned short* __restrict__ uxT, unsigned short* __restrict__ uhT,
                          unsigned short* __restrict__ vxbf, unsigned short* __restrict__ vhT) {
  int t = blockIdx.x * blockDim.x + threadIdx.x;
  if (t < 64 * 1024) {                      // uxT[r][i] = u_x[i][r]
    int r = t >> 10, i = t & 1023;
    uxT[t] = f2bf(u_x[i * 64 + r]);
    return;
  }
  t -= 64 * 1024;
  if (t < 4 * 64 * 512) {                   // uhT[i2g][r][hh] = u_h_i[g][hh][r]
    int i2g = t >> 15, rr = (t >> 9) & 63, hh = t & 511;
    int i = i2g >> 1, g = i2g & 1;
    const float* src = (i == 0) ? u_h_0 : u_h_1;
    uhT[t] = f2bf(src[(g * HG + hh) * 64 + rr]);
    return;
  }
  t -= 4 * 64 * 512;
  if (t < 4096 * 64) {                      // vxbf[n][r] = v_x[n][r] (already [N][K])
    vxbf[t] = f2bf(v_x[t]);
    return;
  }
  t -= 4096 * 64;
  if (t < 4 * 2048 * 64) {                  // vhT[i2g][n][r] = v_h_i[g][r][n]
    int i2g = t >> 17, n = (t >> 6) & 2047, rr = t & 63;
    int i = i2g >> 1, g = i2g & 1;
    const float* src = (i == 0) ? v_h_0 : v_h_1;
    vhT[t] = f2bf(src[(g * 64 + rr) * 2048 + n]);
  }
}

// ---------------- prep: fp32 per-gate diagonals ------------------------------
__global__ void prep_diag(const float* __restrict__ u_x, const float* __restrict__ v_x,
                          const float* __restrict__ u_h_0, const float* __restrict__ v_h_0,
                          float* __restrict__ diag_x, float* __restrict__ diag_h) {
  int t = blockIdx.x * blockDim.x + threadIdx.x;
  if (t < 4096) {                           // diag_x[k][j] = <u_x[j,:], v_x[k*H+j,:]>
    int k = t >> 10, j = t & 1023;
    float s = 0.f;
    for (int r = 0; r < 64; ++r) s += u_x[j * 64 + r] * v_x[(k * H_SZ + j) * 64 + r];
    diag_x[t] = s;
  } else if (t < 8192) {                    // diag_h[k][g*Hg+hh]
    int tt = t - 4096;
    int k = tt >> 10, j = tt & 1023;
    int g = j >> 9, hh = j & 511;
    float s = 0.f;
    for (int r = 0; r < 64; ++r)
      s += u_h_0[(g * HG + hh) * 64 + r] * v_h_0[(g * 64 + r) * 2048 + (k * HG + hh)];
    diag_h[tt] = s;
  }
}

// ---------------- phase 1: rank-64 projections T = A @ U --------------------
// p=0: Tx = x @ u_x (K=1024).  p=1..4 (i2g=p-1): Th[i2g] = h[:, ((g+i)%2)*512:+512] @ u_h_i[g].
__global__ void proj_wmma(const float* __restrict__ x, const float* __restrict__ h,
                          const unsigned short* __restrict__ uxT,
                          const unsigned short* __restrict__ uhT,
                          unsigned short* __restrict__ Tx, unsigned short* __restrict__ Th) {
  int wave = (blockIdx.x * blockDim.x + threadIdx.x) >> 5;
  int lane = threadIdx.x & 31;
  int half = lane >> 4;
  int lrow = lane & 15;
  int p   = wave >> 10;                     // 0..4
  int rem = wave & 1023;
  int tb  = rem >> 2;                       // 0..255
  int tn  = rem & 3;                        // 0..3

  const float* arow;
  const unsigned short* brow;
  unsigned short* out;
  int K;
  if (p == 0) {
    arow = x + (size_t)(tb * 16 + lrow) * H_SZ;
    brow = uxT + (size_t)(tn * 16 + lrow) * 1024;
    K = 1024;
    out = Tx;
  } else {
    int i2g = p - 1;
    int g = i2g & 1, i = i2g >> 1;
    int srcg = (g + i) & 1;                 // cyclic roll of hidden groups
    arow = h + (size_t)(tb * 16 + lrow) * H_SZ + srcg * HG;
    brow = uhT + (size_t)i2g * (64 * 512) + (size_t)(tn * 16 + lrow) * 512;
    K = 512;
    out = Th + (size_t)i2g * ((size_t)B_SZ * 64);
  }

  v8f acc = {};
  for (int kb = 0; kb < K; kb += 32) {
    v16bf a = load_a_f32(arow, half, kb);
    v16bf b = load_b_op(brow, half, kb);
    acc = wmma_bf16(a, b, acc);
  }
  int ncol = tn * 16 + lrow;
#pragma unroll
  for (int e = 0; e < 8; ++e) {
    int m = tb * 16 + e + 8 * half;         // C/D layout: M = e + 8*half, N = lane&15
    out[(size_t)m * 64 + ncol] = f2bf(acc[e]);
  }
}

// ---------------- phase 2: expansion GEMMs + fused gate epilogue ------------
// Block = 8 waves sharing one hidden tile tj; weights for tj staged to LDS via
// global_load_async_to_lds_b128; each wave handles one of 8 batch tiles.
__global__ void __launch_bounds__(256) cell_wmma(
    const float* __restrict__ x, const float* __restrict__ h,
    const float* __restrict__ c, const float* __restrict__ dia_x,
    const float* __restrict__ dia_h, const float* __restrict__ bias_x,
    const float* __restrict__ bias_h,
    const unsigned short* __restrict__ Tx, const unsigned short* __restrict__ Th,
    const unsigned short* __restrict__ vxbf, const unsigned short* __restrict__ vhT,
    const float* __restrict__ diag_x, const float* __restrict__ diag_h,
    float* __restrict__ out_h, float* __restrict__ out_c) {
  // LDS weight cache: rows [s(3)][q(4)][r(16)] of 64 bf16 (K=64) = 192*128B = 24KB
  __shared__ unsigned short smem[192 * 64];

  int tid = threadIdx.x;
  int tj  = blockIdx.x & 63;                // hidden tile (shared by block)
  int tbg = blockIdx.x >> 6;                // batch tile group (8 tiles)
  int g   = tj >> 5;                        // hidden group
  const int kmap[4] = {1, 0, 3, 2};         // gate q (i,f,o,n) -> v_h chunk (f,i,n,o)

  // ---- async stage: 1536 x 16B segments, 6 per thread ----
  const unsigned short* vh0 = vhT + (size_t)(0 * 2 + g) * (2048 * 64);
  const unsigned short* vh1 = vhT + (size_t)(1 * 2 + g) * (2048 * 64);
  unsigned smem_base = (unsigned)(uintptr_t)(&smem[0]);
#pragma unroll
  for (int kk = 0; kk < 6; ++kk) {
    int e   = tid + kk * 256;               // 0..1535
    int row = e >> 3;                       // 0..191
    int seg = e & 7;
    int s   = row >> 6;                     // 0:vx 1:vh0 2:vh1
    int q   = (row >> 4) & 3;
    int r   = row & 15;
    const unsigned short* src;
    if (s == 0)      src = vxbf + (size_t)(q * H_SZ + tj * 16 + r) * 64;
    else if (s == 1) src = vh0 + (size_t)(kmap[q] * HG + (tj & 31) * 16 + r) * 64;
    else             src = vh1 + (size_t)(kmap[q] * HG + (tj & 31) * 16 + r) * 64;
    unsigned lds_off = smem_base + (unsigned)(e * 16);
    const void* gaddr = (const void*)(src + seg * 8);
    asm volatile("global_load_async_to_lds_b128 %0, %1, off"
                 :: "v"(lds_off), "v"(gaddr) : "memory");
  }
  wait_asynccnt0();
  __syncthreads();

  // ---- per-wave 16x16 tile ----
  int wave = tid >> 5;
  int lane = tid & 31;
  int half = lane >> 4;
  int lrow = lane & 15;
  int tb   = tbg * 8 + wave;                // 0..255
  int rowb = tb * 16 + lrow;
  int jcol = tj * 16 + lrow;                // per-lane hidden index

  const unsigned short* txrow  = Tx + (size_t)rowb * 64;
  const unsigned short* th0row = Th + ((size_t)(0 * 2 + g) * ((size_t)B_SZ * 64)) + (size_t)rowb * 64;
  const unsigned short* th1row = Th + ((size_t)(1 * 2 + g) * ((size_t)B_SZ * 64)) + (size_t)rowb * 64;
  v16bf Ax0  = load_a_bf(txrow, half, 0),  Ax1  = load_a_bf(txrow, half, 32);
  v16bf Ah00 = load_a_bf(th0row, half, 0), Ah01 = load_a_bf(th0row, half, 32);
  v16bf Ah10 = load_a_bf(th1row, half, 0), Ah11 = load_a_bf(th1row, half, 32);

  v8f acc[4];
#pragma unroll
  for (int q = 0; q < 4; ++q) {
    const unsigned short* bx  = &smem[(size_t)(0 * 64 + q * 16 + lrow) * 64];
    const unsigned short* bh0 = &smem[(size_t)(1 * 64 + q * 16 + lrow) * 64];
    const unsigned short* bh1 = &smem[(size_t)(2 * 64 + q * 16 + lrow) * 64];
    v8f a = {};
    a = wmma_bf16(Ax0,  load_b_op(bx,  half, 0),  a);
    a = wmma_bf16(Ax1,  load_b_op(bx,  half, 32), a);
    a = wmma_bf16(Ah00, load_b_op(bh0, half, 0),  a);
    a = wmma_bf16(Ah01, load_b_op(bh0, half, 32), a);
    a = wmma_bf16(Ah10, load_b_op(bh1, half, 0),  a);
    a = wmma_bf16(Ah11, load_b_op(bh1, half, 32), a);
    acc[q] = a;
  }

  // j-dependent scalars (uniform over the 8 accumulator rows)
  float dxj = dia_x[jcol], dhj = dia_h[jcol];
  float dgx[4], dgh[4], bxv[4], bhv[4];
#pragma unroll
  for (int k = 0; k < 4; ++k) {
    dgx[k] = diag_x[k * H_SZ + jcol];
    dgh[k] = diag_h[k * H_SZ + jcol];
    bxv[k] = bias_x[k * H_SZ + jcol];
    bhv[k] = bias_h[k * H_SZ + jcol];
  }

#pragma unroll
  for (int e = 0; e < 8; ++e) {
    int b = tb * 16 + e + 8 * half;
    size_t idx = (size_t)b * H_SZ + jcol;
    float xv = x[idx], hv = h[idx], cv = c[idx];
    float base = dxj * xv + dhj * hv;
    float pi = acc[0][e] - xv * dgx[0] + bxv[0] + bhv[1] - hv * dgh[1] + base;
    float pf = acc[1][e] - xv * dgx[1] + bxv[1] + bhv[0] - hv * dgh[0] + base;
    float po = acc[2][e] - xv * dgx[2] + bxv[2] + bhv[3] - hv * dgh[3] + base;
    float pn = acc[3][e] - xv * dgx[3] + bxv[3] + bhv[2] - hv * dgh[2] + base;
    float ig = 1.f / (1.f + __expf(-pi));
    float fg = 1.f / (1.f + __expf(-pf));
    float og = 1.f / (1.f + __expf(-po));
    float e2 = __expf(2.f * pn);
    float ng = (e2 - 1.f) / (e2 + 1.f);     // tanh
    float cn = fg * cv + ig * ng;
    float e2c = __expf(2.f * cn);
    float tc = (e2c - 1.f) / (e2c + 1.f);   // tanh
    out_h[idx] = og * tc;
    out_c[idx] = cn;
  }
}

// ---------------------------------------------------------------------------
extern "C" void kernel_launch(void* const* d_in, const int* in_sizes, int n_in,
                              void* d_out, int out_size, void* d_ws, size_t ws_size,
                              hipStream_t stream) {
  const float* x      = (const float*)d_in[0];
  const float* h      = (const float*)d_in[1];
  const float* c      = (const float*)d_in[2];
  const float* dia_x  = (const float*)d_in[3];
  const float* dia_h  = (const float*)d_in[4];
  const float* u_x    = (const float*)d_in[5];
  const float* v_x    = (const float*)d_in[6];
  const float* u_h_0  = (const float*)d_in[7];
  const float* u_h_1  = (const float*)d_in[8];
  const float* v_h_0  = (const float*)d_in[9];
  const float* v_h_1  = (const float*)d_in[10];
  const float* bias_x = (const float*)d_in[11];
  const float* bias_h = (const float*)d_in[12];
  (void)in_sizes; (void)n_in; (void)out_size; (void)ws_size;

  char* ws = (char*)d_ws;
  size_t off = 0;
  auto take = [&](size_t bytes) -> char* {
    char* p = ws + off;
    off += (bytes + 255) & ~(size_t)255;
    return p;
  };
  unsigned short* Tx    = (unsigned short*)take((size_t)B_SZ * 64 * 2);       // 512 KB
  unsigned short* Th    = (unsigned short*)take((size_t)4 * B_SZ * 64 * 2);   // 2 MB
  unsigned short* uxT   = (unsigned short*)take((size_t)64 * 1024 * 2);       // 128 KB
  unsigned short* uhT   = (unsigned short*)take((size_t)4 * 64 * 512 * 2);    // 256 KB
  unsigned short* vxbf  = (unsigned short*)take((size_t)4096 * 64 * 2);       // 512 KB
  unsigned short* vhT   = (unsigned short*)take((size_t)4 * 2048 * 64 * 2);   // 1 MB
  float*          diagx = (float*)take((size_t)4 * 1024 * 4);                 // 16 KB
  float*          diagh = (float*)take((size_t)4 * 1024 * 4);                 // 16 KB

  // prep: 983040 cast elements, 8192 diag dots
  prep_cast<<<3840, 256, 0, stream>>>(u_x, u_h_0, u_h_1, v_x, v_h_0, v_h_1,
                                      uxT, uhT, vxbf, vhT);
  prep_diag<<<32, 256, 0, stream>>>(u_x, v_x, u_h_0, v_h_0, diagx, diagh);

  // phase 1: 5120 tile-waves (5 projections x 256 x 4), 8 waves/block
  proj_wmma<<<640, 256, 0, stream>>>(x, h, uxT, uhT, Tx, Th);

  // phase 2: 2048 blocks = 64 hidden tiles x 32 batch-tile groups (8 waves each)
  float* out_h = (float*)d_out;
  float* out_c = out_h + (size_t)B_SZ * H_SZ;
  cell_wmma<<<2048, 256, 0, stream>>>(x, h, c, dia_x, dia_h, bias_x, bias_h,
                                      Tx, Th, vxbf, vhT, diagx, diagh, out_h, out_c);
}
